// VGAEEncoder_62852551410248
// MI455X (gfx1250) — matrix operand order repaired and checked by reading, not compiled, not run on hardware
//
#include <hip/hip_runtime.h>

// ---------------------------------------------------------------------------
// VGAE encoder (3x GCNConv) for MI455X / gfx1250, wave32.
//  - GEMMs via v_wmma_f32_16x16x32_f16 (K=96 = 3 wmma steps, M=50000=3125 tiles)
//  - W pre-swizzled into WMMA B-fragment layout in LDS; all B fragments are
//    hoisted into registers once per wave -> no LDS waits inside the MMA chain
//  - Edge gather/scatter via float4 loads + global_atomic_add_f32; node
//    feature buffers (19MB) are L2-resident on the 192MB L2.
// ---------------------------------------------------------------------------

typedef __attribute__((ext_vector_type(16))) _Float16 v16h;
typedef __attribute__((ext_vector_type(8)))  float    v8f;

#define N_NODES 50000
#define N_EDGES 800000
#define M_TILES (N_NODES / 16)   // 3125, exact

// ---------------- degree / normalization ----------------

__global__ void k_init_deg(float* deg, int n) {
    int i = blockIdx.x * blockDim.x + threadIdx.x;
    if (i < n) deg[i] = 1.0f;               // self-loop contributes 1
}

__global__ void k_count_deg(const long long* __restrict__ ei, float* deg, int e) {
    int i = blockIdx.x * blockDim.x + threadIdx.x;
    if (i < e) {
        long long d = ei[e + i];            // dst row of edge_index
        unsafeAtomicAdd(&deg[d], 1.0f);
    }
}

__global__ void k_finish_dinv(float* deg, int n) {
    int i = blockIdx.x * blockDim.x + threadIdx.x;
    if (i < n) deg[i] = rsqrtf(deg[i]);     // in place: deg -> dinv
}

__global__ void k_coef(const long long* __restrict__ ei, const float* __restrict__ dinv,
                       float* coef, int e) {
    int i = blockIdx.x * blockDim.x + threadIdx.x;
    if (i < e) {
        long long s = ei[i];
        long long d = ei[e + i];
        coef[i] = dinv[s] * dinv[d];
    }
}

// ---------------- WMMA GEMM: out[N x FOUT] = act(A[N x 96]) @ W[96 x FOUT] ----------------
// One wave computes a 16-row strip x all FOUT columns. 8 waves / block.

template <int FOUT, bool RELU>
__global__ void __launch_bounds__(256)
k_gemm_wmma(const float* __restrict__ A, const float* __restrict__ W,
            float* __restrict__ out, int nTiles) {
    constexpr int NT = FOUT / 16;                 // n-tiles (6 or 4)
    __shared__ v16h Wfrag[3 * NT * 32];           // [ks][ntile][lane] fragments

    // Stage + swizzle W into this kernel's B-fragment order (contiguous K, same as A): lane l
    // holds column n = l&15; lanes 0-15 carry K=k0..k0+15, lanes 16-31 carry
    // K=k0+16..k0+31; half h -> K = kb + h.
    {
        _Float16* Wh = (_Float16*)Wfrag;
        const int total = 3 * NT * 32 * 16;
        for (int i = threadIdx.x; i < total; i += blockDim.x) {
            int h    = i & 15;
            int lane = (i >> 4) & 31;
            int ft   = i >> 9;                    // 0 .. 3*NT-1
            int ks   = ft / NT;
            int t    = ft - ks * NT;
            int kb   = (lane < 16) ? 0 : 16;
            int k    = ks * 32 + kb + h;
            int n    = t * 16 + (lane & 15);
            Wh[i] = (_Float16)W[k * FOUT + n];
        }
    }
    __syncthreads();

    const int wave = threadIdx.x >> 5;
    const int lane = threadIdx.x & 31;
    const int tile = blockIdx.x * 8 + wave;
    if (tile >= nTiles) return;                   // wave-uniform: EXEC stays all-ones

    // Hoist ALL B fragments into registers: one ds clause, one wait, then the
    // WMMA chain never touches LDS again.
    v16h bfrag[3 * NT];
#pragma unroll
    for (int i = 0; i < 3 * NT; ++i) bfrag[i] = Wfrag[i * 32 + lane];

    const int row0 = tile * 16;
    const int m    = lane & 15;
    const int kbA  = (lane < 16) ? 0 : 16;        // A-fragment K base per lane half
    const float* __restrict__ Arow = A + (size_t)(row0 + m) * 96;

    v8f acc[NT];
#pragma unroll
    for (int t = 0; t < NT; ++t) acc[t] = (v8f){0,0,0,0,0,0,0,0};

#pragma unroll
    for (int ks = 0; ks < 3; ++ks) {              // K = 96 = 3 x 32
        const int k0 = ks * 32;

        // A fragment: 16x32 f16; VGPR j holds K pair (ISA 7.12.2)
        v16h a;
#pragma unroll
        for (int j = 0; j < 8; ++j) {
            const int koff = k0 + kbA + 2 * j;
            float x0 = Arow[koff];
            float x1 = Arow[koff + 1];
            if (RELU) { x0 = fmaxf(x0, 0.0f); x1 = fmaxf(x1, 0.0f); }
            a[2 * j]     = (_Float16)x0;
            a[2 * j + 1] = (_Float16)x1;
        }

#pragma unroll
        for (int t = 0; t < NT; ++t) {
            acc[t] = __builtin_amdgcn_wmma_f32_16x16x32_f16(
                false, a, false, bfrag[ks * NT + t], (short)0, acc[t], false, false);
        }
    }

    // D layout: lanes 0-15 -> N=lane, M=r; lanes 16-31 -> N=lane-16, M=8+r
    const int mrow0 = (lane < 16) ? 0 : 8;
#pragma unroll
    for (int t = 0; t < NT; ++t) {
#pragma unroll
        for (int r = 0; r < 8; ++r) {
            out[(size_t)(row0 + mrow0 + r) * FOUT + t * 16 + m] = acc[t][r];
        }
    }
}

// ---------------- agg init: agg = dinv^2 * h + bias ----------------

__global__ void k_init_agg(const float* __restrict__ h, const float* __restrict__ dinv,
                           const float* __restrict__ bias, float* __restrict__ agg,
                           int total, int F) {
    int idx = blockIdx.x * blockDim.x + threadIdx.x;
    if (idx < total) {
        int i = idx / F;
        int f = idx - i * F;
        float d = dinv[i];
        agg[idx] = d * d * h[idx] + bias[f];
    }
}

// ---------------- edge scatter: agg[dst] += coef[e] * h[src] ----------------
// One thread per (edge, 4-float chunk). Gathers hit L2 (h is 19MB << 192MB L2).

__global__ void k_scatter(const long long* __restrict__ ei, const float* __restrict__ coef,
                          const float* __restrict__ h, float* __restrict__ agg,
                          int e, int F) {
    const int F4 = F >> 2;
    int idx = blockIdx.x * blockDim.x + threadIdx.x;
    if (idx >= e * F4) return;
    int ed = idx / F4;
    int c  = (idx - ed * F4) << 2;
    long long s = ei[ed];
    long long d = ei[e + ed];
    float cf = coef[ed];
    const float4 v = *(const float4*)(h + (size_t)s * F + c);
    float* ap = agg + (size_t)d * F + c;
    unsafeAtomicAdd(ap + 0, v.x * cf);
    unsafeAtomicAdd(ap + 1, v.y * cf);
    unsafeAtomicAdd(ap + 2, v.z * cf);
    unsafeAtomicAdd(ap + 3, v.w * cf);
}

// ---------------- logstd = mu (module aliases them) ----------------

__global__ void k_dup(const float* __restrict__ src, float* __restrict__ dst, int n) {
    int i = blockIdx.x * blockDim.x + threadIdx.x;
    if (i < n) dst[i] = src[i];
}

// ---------------------------------------------------------------------------

static inline int cdiv(long long a, int b) { return (int)((a + b - 1) / b); }

extern "C" void kernel_launch(void* const* d_in, const int* in_sizes, int n_in,
                              void* d_out, int out_size, void* d_ws, size_t ws_size,
                              hipStream_t stream) {
    (void)in_sizes; (void)n_in; (void)out_size; (void)ws_size;

    const float*     x  = (const float*)d_in[0];
    const long long* ei = (const long long*)d_in[1];   // int64 (2, E): src row then dst row
    const float* W0 = (const float*)d_in[2];
    const float* b0 = (const float*)d_in[3];
    const float* W1 = (const float*)d_in[4];
    const float* b1 = (const float*)d_in[5];
    const float* W2 = (const float*)d_in[6];
    const float* b2 = (const float*)d_in[7];
    float* out = (float*)d_out;                        // [mu (N x 64)] [logstd (N x 64)]

    const int N = N_NODES, E = N_EDGES;

    // Workspace layout (floats): dinv | coef | hA (N x 96) | hB (N x 96)
    float* dinv = (float*)d_ws;
    float* coef = dinv + N;
    float* hA   = coef + E;
    float* hB   = hA + (size_t)N * 96;

    const int T = 256;
    const int gemmGrid = cdiv(M_TILES, 8);             // 8 waves (tiles) per block

    // --- normalization coefficients ---
    k_init_deg   <<<cdiv(N, T), T, 0, stream>>>(dinv, N);
    k_count_deg  <<<cdiv(E, T), T, 0, stream>>>(ei, dinv, E);
    k_finish_dinv<<<cdiv(N, T), T, 0, stream>>>(dinv, N);
    k_coef       <<<cdiv(E, T), T, 0, stream>>>(ei, dinv, coef, E);

    // --- layer 0: h = relu(agg(x @ W0) + b0) --- (relu fused into next GEMM's A load)
    k_gemm_wmma<96, false><<<gemmGrid, T, 0, stream>>>(x, W0, hA, M_TILES);
    k_init_agg<<<cdiv((long long)N * 96, T), T, 0, stream>>>(hA, dinv, b0, hB, N * 96, 96);
    k_scatter <<<cdiv((long long)E * 24, T), T, 0, stream>>>(ei, coef, hA, hB, E, 96);

    // --- layer 1 ---
    k_gemm_wmma<96, true><<<gemmGrid, T, 0, stream>>>(hB, W1, hA, M_TILES);
    k_init_agg<<<cdiv((long long)N * 96, T), T, 0, stream>>>(hA, dinv, b1, hB, N * 96, 96);
    k_scatter <<<cdiv((long long)E * 24, T), T, 0, stream>>>(ei, coef, hA, hB, E, 96);

    // --- layer 2 (mu, no relu) --- writes straight into d_out's mu half
    k_gemm_wmma<64, true><<<gemmGrid, T, 0, stream>>>(hB, W2, hA, M_TILES);
    k_init_agg<<<cdiv((long long)N * 64, T), T, 0, stream>>>(hA, dinv, b2, out, N * 64, 64);
    k_scatter <<<cdiv((long long)E * 16, T), T, 0, stream>>>(ei, coef, hA, out, E, 64);

    // logstd = mu
    k_dup<<<cdiv((long long)N * 64, T), T, 0, stream>>>(out, out + (size_t)N * 64, N * 64);
}
